// Attention_59236188946589
// MI455X (gfx1250) — matrix operand order repaired and checked by reading, not compiled, not run on hardware
//
#include <hip/hip_runtime.h>
#include <hip/hip_bf16.h>
#include <stdint.h>

#define BATCH 4
#define NTOK  4096
#define CHN   256
#define DQK   32

#define VROW  72   // padded LDS row (elems) for V^T tile: 144B = 36 banks -> conflict-free frag reads
#define KROW  40   // padded LDS row for K tile: 80B = 20 banks
#define PROW  72   // padded LDS row for P tile

#if __has_builtin(__builtin_amdgcn_global_load_async_to_lds_b128)
#define HAVE_ASYNC_LDS 1
#endif

typedef __attribute__((ext_vector_type(16))) __bf16 v16bf;
typedef __attribute__((ext_vector_type(8)))  float  v8f;

typedef int v4i __attribute__((vector_size(16)));
typedef __attribute__((address_space(1))) v4i gv4i;   // global (prints as __device__)
typedef __attribute__((address_space(3))) v4i lv4i;   // LDS    (prints as __shared__)

union FragU {
    struct { uint4 lo, hi; } q;
    v16bf v;
};

__device__ __forceinline__ unsigned short f2bf(float f) {
    unsigned int u = __float_as_uint(f);
    unsigned int r = (u + 0x7FFFu + ((u >> 16) & 1u)) >> 16;  // round-to-nearest-even
    return (unsigned short)r;
}

// 16-byte chunk copy global -> LDS. Async (ASYNCcnt path) when available.
__device__ __forceinline__ void copy16(const void* g, void* l) {
#ifdef HAVE_ASYNC_LDS
    __builtin_amdgcn_global_load_async_to_lds_b128((gv4i*)g, (lv4i*)l, 0, 0);
#else
    *(uint4*)l = *(const uint4*)g;
#endif
}
__device__ __forceinline__ void wait_async_copies() {
#ifdef HAVE_ASYNC_LDS
#if __has_builtin(__builtin_amdgcn_s_wait_asynccnt)
    __builtin_amdgcn_s_wait_asynccnt(0);
#else
    asm volatile("s_wait_asynccnt 0" ::: "memory");
#endif
#endif
}

// ---- DPP butterfly reductions within each 16-lane half ----
__device__ __forceinline__ float dpp_max16(float v) {
    int y;
    y = __builtin_amdgcn_update_dpp(0, __float_as_int(v), 0xB1, 0xf, 0xf, true);  // quad_perm [1,0,3,2]
    v = fmaxf(v, __int_as_float(y));
    y = __builtin_amdgcn_update_dpp(0, __float_as_int(v), 0x4E, 0xf, 0xf, true);  // quad_perm [2,3,0,1]
    v = fmaxf(v, __int_as_float(y));
    y = __builtin_amdgcn_update_dpp(0, __float_as_int(v), 0x141, 0xf, 0xf, true); // row_half_mirror
    v = fmaxf(v, __int_as_float(y));
    y = __builtin_amdgcn_update_dpp(0, __float_as_int(v), 0x140, 0xf, 0xf, true); // row_mirror
    v = fmaxf(v, __int_as_float(y));
    return v;
}
__device__ __forceinline__ float dpp_sum16(float v) {
    int y;
    y = __builtin_amdgcn_update_dpp(0, __float_as_int(v), 0xB1, 0xf, 0xf, true);
    v += __int_as_float(y);
    y = __builtin_amdgcn_update_dpp(0, __float_as_int(v), 0x4E, 0xf, 0xf, true);
    v += __int_as_float(y);
    y = __builtin_amdgcn_update_dpp(0, __float_as_int(v), 0x141, 0xf, 0xf, true);
    v += __int_as_float(y);
    y = __builtin_amdgcn_update_dpp(0, __float_as_int(v), 0x140, 0xf, 0xf, true);
    v += __int_as_float(y);
    return v;
}

// A fragment (16xK, K>=32, bf16, row-major memory or LDS).
__device__ __forceinline__ v16bf load_a_frag(const unsigned short* base, int stride_elems, int lane) {
    const int m    = lane & 15;
    const int half = lane >> 4;
    const char* row = (const char*)(base + (size_t)m * stride_elems);
    FragU u;
    u.q.lo = *(const uint4*)(row + half * 16);
    u.q.hi = *(const uint4*)(row + 32 + half * 16);
    return u.v;
}

// B fragment (32x16 KxN); memory holds B^T row-major with >=32 columns.
__device__ __forceinline__ v16bf load_b_frag(const unsigned short* baseT, size_t stride_elems,
                                             int lane, int k_off) {
    const int n    = lane & 15;
    const int koff = k_off + ((lane >> 4) << 3);
    const char* row = (const char*)(baseT + (size_t)n * stride_elems);
    FragU u;
    u.q.lo = *(const uint4*)(row + koff * 2);
    u.q.hi = *(const uint4*)(row + koff * 2 + 32);
    return u.v;
}

// ---------------- P1: x -> bf16; f,g projections ----------------
__global__ __launch_bounds__(256) void prep_kernel(
    const float* __restrict__ x, const float* __restrict__ Wf, const float* __restrict__ Wg,
    const float* __restrict__ bf, const float* __restrict__ bg,
    unsigned short* __restrict__ xbf, unsigned short* __restrict__ Qb, unsigned short* __restrict__ Kb)
{
    __shared__ float sx[CHN];
    const int tok = blockIdx.x;
    const int c   = threadIdx.x;
    const float v = x[(size_t)tok * CHN + c];
    sx[c] = v;
    xbf[(size_t)tok * CHN + c] = f2bf(v);
    __syncthreads();
    if (c < 2 * DQK) {
        const int d = c & (DQK - 1);
        const float* W = (c < DQK) ? Wf : Wg;
        float acc = (c < DQK) ? bf[d] : bg[d];
#pragma unroll 8
        for (int i = 0; i < CHN; ++i) acc = fmaf(sx[i], W[i * DQK + d], acc);
        if (c < DQK) Kb[(size_t)tok * DQK + d] = f2bf(acc);
        else         Qb[(size_t)tok * DQK + d] = f2bf(acc);
    }
}

// ---------------- P0: Wh^T bf16 [C_out][C_in] ----------------
__global__ __launch_bounds__(256) void wtrans_kernel(const float* __restrict__ Wh,
                                                     unsigned short* __restrict__ WhT)
{
    const int ci = blockIdx.x;
    const int co = threadIdx.x;
    WhT[(size_t)co * CHN + ci] = f2bf(Wh[(size_t)ci * CHN + co]);
}

// ---------------- P2: h = x*Wh + bh via WMMA, stored as V^T bf16 [B][C][N] ----------------
__global__ __launch_bounds__(256) void hgemm_kernel(
    const unsigned short* __restrict__ xbf, const unsigned short* __restrict__ WhT,
    const float* __restrict__ bh, unsigned short* __restrict__ VT)
{
    const int lane  = threadIdx.x & 31;
    const int w     = threadIdx.x >> 5;
    const int col   = lane & 15;
    const int half  = lane >> 4;
    const int tok0  = blockIdx.x * 64 + (w & 3) * 16;
    const int chalf = w >> 2;
    const int bidx  = tok0 >> 12;
    const int n0    = tok0 & (NTOK - 1);

    const v8f vzero = {0.f, 0.f, 0.f, 0.f, 0.f, 0.f, 0.f, 0.f};
    v16bf a[8];
#pragma unroll
    for (int kg = 0; kg < 8; ++kg)
        a[kg] = load_a_frag(xbf + (size_t)tok0 * CHN + kg * 32, CHN, lane);

#pragma unroll
    for (int ct = 0; ct < 8; ++ct) {
        const int ch0 = chalf * 128 + ct * 16;
        v8f acc = vzero;
#pragma unroll
        for (int kg = 0; kg < 8; ++kg) {
            v16bf bfr = load_b_frag(WhT + (size_t)ch0 * CHN, CHN, lane, kg * 32);
            acc = __builtin_amdgcn_wmma_f32_16x16x32_bf16(false, a[kg], false, bfr,
                                                          (short)0, acc, false, false);
        }
        const int ch = ch0 + col;
        const float bias = bh[ch];
        unsigned short* vtrow = VT + ((size_t)bidx * CHN + ch) * NTOK;
#pragma unroll
        for (int j = 0; j < 8; j += 2) {
            unsigned int pk = (unsigned int)f2bf(acc[j] + bias)
                            | ((unsigned int)f2bf(acc[j + 1] + bias) << 16);
            *(unsigned int*)(vtrow + n0 + 8 * half + j) = pk;
        }
    }
}

// ---------------- Flash attention with double-buffered async LDS staging ----------------
// Block = 4 waves = 64 queries; per iter stage K tile [64][32] and V^T tile [256][64] to LDS.
__global__ __launch_bounds__(128) void flash_kernel(
    const unsigned short* __restrict__ Qb, const unsigned short* __restrict__ Kb,
    const unsigned short* __restrict__ VT, const float* __restrict__ x,
    const float* __restrict__ gamma, float* __restrict__ out)
{
    __shared__ __align__(16) unsigned short Vs[2][CHN * VROW];
    __shared__ __align__(16) unsigned short Ks[2][64 * KROW];
    __shared__ __align__(16) unsigned short Plds[4][16 * PROW];

    const int tid  = threadIdx.x;
    const int lane = tid & 31;
    const int w    = tid >> 5;
    const int bidx = blockIdx.y;
    const int q0   = blockIdx.x * 64 + w * 16;
    const int col  = lane & 15;
    const int half = lane >> 4;

    const char* Vg = (const char*)(VT + (size_t)bidx * CHN * NTOK);   // rows of NTOK bf16
    const char* Kg = (const char*)(Kb + (size_t)bidx * NTOK * DQK);

    auto stage = [&](int buf, int kb) {
        // V^T tile: 256 rows x 128B, 2048 x 16B chunks, 16 per thread
        unsigned short* vd = &Vs[buf][0];
        const char* vsrc = Vg + (size_t)kb * 2;
#pragma unroll
        for (int i = 0; i < 16; ++i) {
            const int c  = i * 128 + tid;
            const int ch = c >> 3;
            const int of = (c & 7) * 16;
            copy16(vsrc + (size_t)ch * (NTOK * 2) + of, (char*)(vd + ch * VROW) + of);
        }
        // K tile: 64 rows x 64B, 256 x 16B chunks, 2 per thread
        unsigned short* kd = &Ks[buf][0];
        const char* ksrc = Kg + (size_t)kb * (DQK * 2);
#pragma unroll
        for (int i = 0; i < 2; ++i) {
            const int c  = i * 128 + tid;
            const int r  = c >> 2;
            const int of = (c & 3) * 16;
            copy16(ksrc + r * (DQK * 2) + of, (char*)(kd + r * KROW) + of);
        }
    };

    const v16bf qf = load_a_frag(Qb + ((size_t)bidx * NTOK + q0) * DQK, DQK, lane);

    const v8f vzero = {0.f, 0.f, 0.f, 0.f, 0.f, 0.f, 0.f, 0.f};
    v8f   O[16];
    float m_i[8], l_i[8];
#pragma unroll
    for (int t = 0; t < 16; ++t) O[t] = vzero;
#pragma unroll
    for (int j = 0; j < 8; ++j) { m_i[j] = -1e30f; l_i[j] = 0.f; }

    unsigned short* myP = &Plds[w][0];

    stage(0, 0);

    for (int it = 0; it < NTOK / 64; ++it) {
        const int buf = it & 1;
        wait_async_copies();
        __syncthreads();                       // tile `buf` ready; prior reads of buf^1 done
        if (it + 1 < NTOK / 64) stage(buf ^ 1, (it + 1) * 64);

        // ---- S = Q * K^T from LDS K tile
        v8f S[4];
#pragma unroll
        for (int t = 0; t < 4; ++t) {
            v16bf kf = load_b_frag(&Ks[buf][t * 16 * KROW], KROW, lane, 0);
            S[t] = __builtin_amdgcn_wmma_f32_16x16x32_bf16(false, qf, false, kf,
                                                           (short)0, vzero, false, false);
        }
        // ---- online softmax
        float corr[8];
#pragma unroll
        for (int j = 0; j < 8; ++j) {
            float mx = fmaxf(fmaxf(S[0][j], S[1][j]), fmaxf(S[2][j], S[3][j]));
            mx = dpp_max16(mx);
            const float mn = fmaxf(m_i[j], mx);
            corr[j] = __expf(m_i[j] - mn);
            m_i[j]  = mn;
            float rs = 0.f;
#pragma unroll
            for (int t = 0; t < 4; ++t) { float p = __expf(S[t][j] - mn); S[t][j] = p; rs += p; }
            rs = dpp_sum16(rs);
            l_i[j] = l_i[j] * corr[j] + rs;
        }
#pragma unroll
        for (int ct = 0; ct < 16; ++ct)
#pragma unroll
            for (int j = 0; j < 8; ++j) O[ct][j] *= corr[j];

        // ---- P: regs -> per-wave LDS -> A-layout fragments
#pragma unroll
        for (int t = 0; t < 4; ++t)
#pragma unroll
            for (int j = 0; j < 8; ++j)
                myP[(j + 8 * half) * PROW + t * 16 + col] = f2bf(S[t][j]);
        asm volatile("s_wait_dscnt 0" ::: "memory");   // DS in-order within wave
        const v16bf p0 = load_a_frag(myP,      PROW, lane);
        const v16bf p1 = load_a_frag(myP + 32, PROW, lane);

        // ---- O += P * V from LDS V^T tile
#pragma unroll
        for (int ct = 0; ct < 16; ++ct) {
            const unsigned short* vrow = &Vs[buf][ct * 16 * VROW];
            v16bf b0 = load_b_frag(vrow, VROW, lane, 0);
            O[ct] = __builtin_amdgcn_wmma_f32_16x16x32_bf16(false, p0, false, b0,
                                                            (short)0, O[ct], false, false);
            v16bf b1 = load_b_frag(vrow, VROW, lane, 32);
            O[ct] = __builtin_amdgcn_wmma_f32_16x16x32_bf16(false, p1, false, b1,
                                                            (short)0, O[ct], false, false);
        }
    }

    // ---- epilogue: out = gamma * (O / l) + x
    const float gm = gamma[0];
    float inv[8];
#pragma unroll
    for (int j = 0; j < 8; ++j) inv[j] = 1.0f / l_i[j];
#pragma unroll
    for (int ct = 0; ct < 16; ++ct) {
#pragma unroll
        for (int j = 0; j < 8; ++j) {
            const int m = j + 8 * half;
            const size_t idx = ((size_t)bidx * NTOK + q0 + m) * CHN + ct * 16 + col;
            out[idx] = gm * (O[ct][j] * inv[j]) + x[idx];
        }
    }
}

extern "C" void kernel_launch(void* const* d_in, const int* in_sizes, int n_in,
                              void* d_out, int out_size, void* d_ws, size_t ws_size,
                              hipStream_t stream) {
    (void)in_sizes; (void)n_in; (void)out_size; (void)ws_size;
    const float* x     = (const float*)d_in[0];
    const float* Wf    = (const float*)d_in[1];
    const float* Wg    = (const float*)d_in[2];
    const float* Wh    = (const float*)d_in[3];
    const float* bf    = (const float*)d_in[4];
    const float* bg    = (const float*)d_in[5];
    const float* bh    = (const float*)d_in[6];
    const float* gamma = (const float*)d_in[7];
    float* out = (float*)d_out;

    unsigned short* Qb  = (unsigned short*)d_ws;
    unsigned short* Kb  = Qb  + (size_t)BATCH * NTOK * DQK;
    unsigned short* VT  = Kb  + (size_t)BATCH * NTOK * DQK;
    unsigned short* xbf = VT  + (size_t)BATCH * CHN * NTOK;
    unsigned short* WhT = xbf + (size_t)BATCH * NTOK * CHN;

    prep_kernel  <<<dim3(BATCH * NTOK),      dim3(256), 0, stream>>>(x, Wf, Wg, bf, bg, xbf, Qb, Kb);
    wtrans_kernel<<<dim3(CHN),               dim3(256), 0, stream>>>(Wh, WhT);
    hgemm_kernel <<<dim3(BATCH * NTOK / 64), dim3(256), 0, stream>>>(xbf, WhT, bh, VT);
    flash_kernel <<<dim3(NTOK / 64, BATCH),  dim3(128), 0, stream>>>(Qb, Kb, VT, x, gamma, out);
}
